// MyNet_11879879543557
// MI455X (gfx1250) — matrix operand order repaired and checked by reading, not compiled, not run on hardware
//
#include <hip/hip_runtime.h>
#include <hip/hip_bf16.h>

// ---------------------------------------------------------------------------
// out[b,t,i,j] = delta_ij * s[b,t],  s[b,t] = d y[t]/d u[t],
// u[b,t] = trace(x[b,t]), y = Linear(LSTM1(LSTM0(u))).  B=8, T=128, n=32, H=128.
// One workgroup, 32 wave32 waves; batches packed in M=16 of WMMA tiles.
// Weight placement (128-VGPR/wave budget at 1024 threads):
//   - w_ih1, w_hh1 fragments: register-resident (64 VGPRs/lane)
//   - w_hh0: staged in LDS (128 KB) and read as B-fragments per step
// 4 barriers per step; branchless s_out drain (no loop peel).
// ---------------------------------------------------------------------------

typedef __attribute__((ext_vector_type(16))) _Float16 v16h;
typedef __attribute__((ext_vector_type(8)))  _Float16 v8h;
typedef __attribute__((ext_vector_type(8)))  float    v8f;

__device__ __forceinline__ float sigm(float x) { return 1.0f / (1.0f + __expf(-x)); }

__device__ __forceinline__ v16h pack16(const _Float16* p0, const _Float16* p1) {
    v8h a = *(const v8h*)p0;
    v8h b = *(const v8h*)p1;
    v16h r;
#pragma unroll
    for (int i = 0; i < 8; ++i) { r[i] = a[i]; r[i + 8] = b[i]; }
    return r;
}

// A (16x32 f16) fragment from LDS h-state [16][128]:
// lane L: row L%16; halves 0..7 = K kb+8*hi, halves 8..15 = K kb+16+8*hi.
__device__ __forceinline__ v16h loadA(const _Float16* hbuf, int row, int kb, int hi) {
    const _Float16* base = hbuf + row * 128 + kb;
    return pack16(base + hi * 8, base + 16 + hi * 8);
}

// B (32x16 f16) fragment from weights [512][128] row-major (global or LDS):
// lane L: col L%16 (weight row nrow); halves 0..7 = K kb+8*hi, halves 8..15 = K kb+16+8*hi.
__device__ __forceinline__ v16h loadB(const _Float16* w, int nrow, int kb, int hi) {
    const _Float16* base = w + (size_t)nrow * 128 + kb;
    return pack16(base + hi * 8, base + 16 + hi * 8);
}

__device__ __forceinline__ v8f wmma_f16(v16h a, v16h b, v8f c) {
    return __builtin_amdgcn_wmma_f32_16x16x32_f16(false, a, false, b, (short)0, c,
                                                  false, false);
}

// ---------------------------------------------------------------------------
// Prep kernels
// ---------------------------------------------------------------------------
__global__ void cvt_f32_to_f16(const float* __restrict__ src,
                               _Float16* __restrict__ dst, int n) {
    int i = blockIdx.x * blockDim.x + threadIdx.x;
    if (i < n) dst[i] = (_Float16)src[i];
}

__global__ void trace_kernel(const float* __restrict__ x, float* __restrict__ u, int nBT) {
    int e = blockIdx.x * blockDim.x + threadIdx.x;
    if (e < nBT) {
        const float* p = x + (size_t)e * 1024;  // one 32x32 matrix
        float s = 0.0f;
#pragma unroll
        for (int i = 0; i < 32; ++i) s += p[i * 33];
        u[e] = s;
    }
}

__global__ void out_kernel(const float* __restrict__ s, float* __restrict__ out,
                           int total) {
    int idx = blockIdx.x * blockDim.x + threadIdx.x;
    if (idx < total) {
        int bt = idx >> 10;
        int r  = idx & 1023;
        int i  = r >> 5, j = r & 31;
        out[idx] = (i == j) ? s[bt] : 0.0f;
    }
}

// ---------------------------------------------------------------------------
// Sequential scan: 1024 threads = 32 waves, wave w owns gate columns
// [w*16, w*16+16).  16 WMMA per wave per step.
// ---------------------------------------------------------------------------
__global__ __launch_bounds__(1024) void lstm_scan_kernel(
    const _Float16* __restrict__ w_hh0_h,   // [512][128] f16
    const _Float16* __restrict__ w_ih1_h,   // [512][128] f16
    const _Float16* __restrict__ w_hh1_h,   // [512][128] f16
    const float* __restrict__ u_g,          // [8][128]
    const float* __restrict__ w_ih0,        // [512]
    const float* __restrict__ b_ih0, const float* __restrict__ b_hh0,
    const float* __restrict__ b_ih1, const float* __restrict__ b_hh1,
    const float* __restrict__ w_out,        // [128]
    float* __restrict__ s_out)              // [8][128]
{
    __shared__ __align__(16) _Float16 w0L[512 * 128]; // w_hh0 staged in LDS (128 KB)
    __shared__ __align__(16) _Float16 hA[16][128];    // layer0 h (= layer1 input)
    __shared__ __align__(16) _Float16 h2A[16][128];   // layer1 h
    __shared__ __align__(16) _Float16 dh1A[16][128];  // tangent dh1
    __shared__ float gF[8][512];                      // forward gate pre-acts (raw)
    __shared__ float gT[8][512];                      // tangent gate pre-acts
    __shared__ float c0s[8][128];
    __shared__ float c1s[8][128];
    __shared__ float uL[8][128];
    __shared__ float bsum0[512];                      // b_ih0 + b_hh0
    __shared__ float bsum1[512];                      // b_ih1 + b_hh1
    __shared__ float wiL[512];                        // w_ih0
    __shared__ float woL[128];                        // w_out
    __shared__ float sbuf[8];

    const int tid  = threadIdx.x;
    const int wave = tid >> 5;          // 0..31 -> N-tile index
    const int lane = tid & 31;
    const int L2   = lane & 15;
    const int hi   = lane >> 4;
    const int ncol = wave * 16 + L2;    // gate column this lane owns in C/D

    // ---- stage w_hh0 into LDS (8192 x v8h chunks, 8 per thread) ----
    {
        const v8h* src = (const v8h*)w_hh0_h;
        v8h*       dst = (v8h*)w0L;
        for (int e = tid; e < 512 * 128 / 8; e += 1024) dst[e] = src[e];
    }

    // ---- init state (padding rows 8..15 stay zero forever) ----
    for (int e = tid; e < 16 * 128; e += 1024) {
        (&hA[0][0])[e]   = (_Float16)0.0f;
        (&h2A[0][0])[e]  = (_Float16)0.0f;
        (&dh1A[0][0])[e] = (_Float16)0.0f;
    }
    if (tid < 8 * 128) {
        (&c0s[0][0])[tid] = 0.0f;
        (&c1s[0][0])[tid] = 0.0f;
        (&uL[0][0])[tid]  = u_g[tid];
    }
    if (tid < 512) {
        bsum0[tid] = b_ih0[tid] + b_hh0[tid];
        bsum1[tid] = b_ih1[tid] + b_hh1[tid];
        wiL[tid]   = w_ih0[tid];
    }
    if (tid < 128) woL[tid] = w_out[tid];
    if (tid < 8)   sbuf[tid] = 0.0f;

    // ---- preload layer-1 weight B-fragments into registers (64 VGPRs) ----
    v16h Bih1[4], Bhh1[4];
#pragma unroll
    for (int kk = 0; kk < 4; ++kk) {
        Bih1[kk] = loadB(w_ih1_h, ncol, kk * 32, hi);
        Bhh1[kk] = loadB(w_hh1_h, ncol, kk * 32, hi);
    }

    // per-thread elementwise coordinates (1024 items: b in 0..7, j in 0..127)
    const int eb = tid >> 7;
    const int ej = tid & 127;

    __syncthreads();

    for (int t = 0; t < 128; ++t) {
        // ---- phase A: layer-0 recurrent matmul (B-frags from LDS) ----
        v8f acc0 = {};
#pragma unroll
        for (int kk = 0; kk < 4; ++kk) {
            v16h B = loadB(w0L, ncol, kk * 32, hi);
            v16h A = loadA(&hA[0][0], L2, kk * 32, hi);
            acc0 = wmma_f16(A, B, acc0);
        }
        if (hi == 0) {  // rows m=0..7 are the real batches
#pragma unroll
            for (int r = 0; r < 8; ++r)
                gF[r][ncol] = acc0[r];
        }
        if (tid < 8) {  // branchless drain of previous step's reduction
            s_out[tid * 128 + ((t + 127) & 127)] = sbuf[tid];
            sbuf[tid] = 0.0f;
        }
        __syncthreads();

        // ---- phase B: layer-0 elementwise (bias + u*w_ih0 added here) ----
        {
            float ut = uL[eb][t];
            float gi = gF[eb][ej]       + bsum0[ej]       + ut * wiL[ej];
            float gf = gF[eb][128 + ej] + bsum0[128 + ej] + ut * wiL[128 + ej];
            float gg = gF[eb][256 + ej] + bsum0[256 + ej] + ut * wiL[256 + ej];
            float go = gF[eb][384 + ej] + bsum0[384 + ej] + ut * wiL[384 + ej];
            float cp = c0s[eb][ej];
            float iv = sigm(gi), fv = sigm(gf), gv = tanhf(gg), ov = sigm(go);
            float c  = fv * cp + iv * gv;
            c0s[eb][ej] = c;
            float tc = tanhf(c);
            hA[eb][ej] = (_Float16)(ov * tc);
            // same-step JVP, seed dgates0/du = w_ih0
            float di = iv * (1.0f - iv) * wiL[ej];
            float df = fv * (1.0f - fv) * wiL[128 + ej];
            float dg = (1.0f - gv * gv) * wiL[256 + ej];
            float dov = ov * (1.0f - ov) * wiL[384 + ej];
            float dc  = df * cp + di * gv + iv * dg;
            float dh1 = dov * tc + ov * (1.0f - tc * tc) * dc;
            dh1A[eb][ej] = (_Float16)dh1;
        }
        __syncthreads();

        // ---- phase C: layer-1 fwd (h1@w_ih1^T + h2@w_hh1^T) and tangent
        //      (dh1@w_ih1^T), sharing the register-resident Bih1 frags ----
        v8f accF = {}, accT = {};
#pragma unroll
        for (int kk = 0; kk < 4; ++kk) {
            v16h Af = loadA(&hA[0][0], L2, kk * 32, hi);
            accF = wmma_f16(Af, Bih1[kk], accF);
            v16h At = loadA(&dh1A[0][0], L2, kk * 32, hi);
            accT = wmma_f16(At, Bih1[kk], accT);
        }
#pragma unroll
        for (int kk = 0; kk < 4; ++kk) {
            v16h Ah = loadA(&h2A[0][0], L2, kk * 32, hi);
            accF = wmma_f16(Ah, Bhh1[kk], accF);
        }
        if (hi == 0) {
#pragma unroll
            for (int r = 0; r < 8; ++r) {
                gF[r][ncol] = accF[r];
                gT[r][ncol] = accT[r];
            }
        }
        __syncthreads();

        // ---- phase D: layer-1 elementwise + JVP, s[b] = w_out . dh2 ----
        {
            float gi = gF[eb][ej]       + bsum1[ej];
            float gf = gF[eb][128 + ej] + bsum1[128 + ej];
            float gg = gF[eb][256 + ej] + bsum1[256 + ej];
            float go = gF[eb][384 + ej] + bsum1[384 + ej];
            float ti = gT[eb][ej],       tf = gT[eb][128 + ej];
            float tg = gT[eb][256 + ej], to = gT[eb][384 + ej];
            float cp = c1s[eb][ej];
            float iv = sigm(gi), fv = sigm(gf), gv = tanhf(gg), ov = sigm(go);
            float c  = fv * cp + iv * gv;
            c1s[eb][ej] = c;
            float tc = tanhf(c);
            h2A[eb][ej] = (_Float16)(ov * tc);
            float di = iv * (1.0f - iv) * ti;
            float df = fv * (1.0f - fv) * tf;
            float dg = (1.0f - gv * gv) * tg;
            float dov = ov * (1.0f - ov) * to;
            float dc  = df * cp + di * gv + iv * dg;
            float dh2 = dov * tc + ov * (1.0f - tc * tc) * dc;
            atomicAdd(&sbuf[eb], woL[ej] * dh2);   // ds_add_f32
        }
        __syncthreads();
    }
    // drain final step (also fixes the dummy t=0 write to index 127)
    if (tid < 8) s_out[tid * 128 + 127] = sbuf[tid];
}

// ---------------------------------------------------------------------------
// Host-side launcher
// ---------------------------------------------------------------------------
extern "C" void kernel_launch(void* const* d_in, const int* in_sizes, int n_in,
                              void* d_out, int out_size, void* d_ws, size_t ws_size,
                              hipStream_t stream) {
    const float* x      = (const float*)d_in[0];   // [8,128,32,32]
    const float* w_ih0  = (const float*)d_in[1];   // [512,1]
    const float* w_hh0  = (const float*)d_in[2];   // [512,128]
    const float* b_ih0  = (const float*)d_in[3];
    const float* b_hh0  = (const float*)d_in[4];
    const float* w_ih1  = (const float*)d_in[5];   // [512,128]
    const float* w_hh1  = (const float*)d_in[6];   // [512,128]
    const float* b_ih1  = (const float*)d_in[7];
    const float* b_hh1  = (const float*)d_in[8];
    const float* w_out  = (const float*)d_in[9];   // [1,128]
    float* out = (float*)d_out;

    // workspace layout
    char* ws = (char*)d_ws;
    _Float16* w_hh0_h = (_Float16*)(ws + 0);        // 131072 B
    _Float16* w_ih1_h = (_Float16*)(ws + 131072);   // 131072 B
    _Float16* w_hh1_h = (_Float16*)(ws + 262144);   // 131072 B
    float*    u_buf   = (float*)(ws + 393216);      // 4096 B  [8][128]
    float*    s_buf   = (float*)(ws + 397312);      // 4096 B  [8][128]

    const int NW = 512 * 128;  // 65536 weights per matrix
    cvt_f32_to_f16<<<(NW + 255) / 256, 256, 0, stream>>>(w_hh0, w_hh0_h, NW);
    cvt_f32_to_f16<<<(NW + 255) / 256, 256, 0, stream>>>(w_ih1, w_ih1_h, NW);
    cvt_f32_to_f16<<<(NW + 255) / 256, 256, 0, stream>>>(w_hh1, w_hh1_h, NW);

    trace_kernel<<<4, 256, 0, stream>>>(x, u_buf, 1024);

    lstm_scan_kernel<<<1, 1024, 0, stream>>>(w_hh0_h, w_ih1_h, w_hh1_h, u_buf,
                                             w_ih0, b_ih0, b_hh0, b_ih1, b_hh1,
                                             w_out, s_buf);

    const int total = 8 * 128 * 32 * 32;  // 1048576
    out_kernel<<<total / 256, 256, 0, stream>>>(s_buf, out, total);
}
